// TLGv4DecoderLayer_16793322127556
// MI455X (gfx1250) — matrix operand order repaired and checked by reading, not compiled, not run on hardware
//
#include <hip/hip_runtime.h>
#include <hip/hip_bf16.h>
#include <cstdint>

// ---------------- CDNA5 WMMA types ----------------
typedef __bf16 v16bf __attribute__((ext_vector_type(16)));
typedef float  v8f   __attribute__((ext_vector_type(8)));

union AF { v16bf v; uint32_t u[8]; };

__device__ __forceinline__ unsigned short f2bf(float f) {
  uint32_t u = __float_as_uint(f);
  u += 0x7fffu + ((u >> 16) & 1u);          // round-to-nearest-even
  return (unsigned short)(u >> 16);
}

__device__ __forceinline__ bool is_inf(float f) {
  return (__float_as_uint(f) & 0x7fffffffu) == 0x7f800000u;
}

// A-fragment (16x32 bf16, MxK): lane holds row M=lane&15.
// lanes 0-15: K = 0..15 ; lanes 16-31: K = 16..31.
// Two 16B loads (works for global or LDS-backed pointers).
__device__ __forceinline__ AF load_afrag(const unsigned short* rowptr, int koff, int lane) {
  AF f;
  const int kb = (lane & 16) ? 16 : 0;
  const uint4 a = *reinterpret_cast<const uint4*>(rowptr + koff + kb);
  const uint4 b = *reinterpret_cast<const uint4*>(rowptr + koff + kb + 8);
  f.u[0] = a.x; f.u[1] = a.y; f.u[2] = a.z; f.u[3] = a.w;
  f.u[4] = b.x; f.u[5] = b.y; f.u[6] = b.z; f.u[7] = b.w;
  return f;
}

// B-fragment (32x16 bf16, KxN) from LDS stored [n][k] (k contiguous):
// lane holds column N=lane&15; lanes 0-15: K 0..15, lanes 16-31: K 16..31.
__device__ __forceinline__ AF load_bfrag(const unsigned short* base, int row0,
                                         int rowstride, int koff, int lane) {
  AF f;
  const int row = row0 + (lane & 15);
  const int kb  = (lane & 16) ? 16 : 0;
  const uint32_t* p = reinterpret_cast<const uint32_t*>(base + row * rowstride + koff + kb);
#pragma unroll
  for (int i = 0; i < 8; i++) f.u[i] = p[i];
  return f;
}

__device__ __forceinline__ v8f wmma_bf16(const AF& a, const AF& b, v8f c) {
  return __builtin_amdgcn_wmma_f32_16x16x32_bf16(false, a.v, false, b.v,
                                                 (short)0, c, false, false);
}

// ---------------- LayerNorm -> bf16 ----------------
__global__ __launch_bounds__(256) void ln_bf16(const float* __restrict__ x,
                                               const float* __restrict__ w,
                                               const float* __restrict__ b,
                                               unsigned short* __restrict__ out) {
  __shared__ float r1[256], r2[256];
  const int tok = blockIdx.x, tid = threadIdx.x;
  const float* row = x + (size_t)tok * 4096;
  float s = 0.f, ss = 0.f;
#pragma unroll
  for (int j = 0; j < 16; j++) { float v = row[tid + j * 256]; s += v; ss += v * v; }
  r1[tid] = s; r2[tid] = ss;
  __syncthreads();
  for (int off = 128; off > 0; off >>= 1) {
    if (tid < off) { r1[tid] += r1[tid + off]; r2[tid] += r2[tid + off]; }
    __syncthreads();
  }
  const float mean = r1[0] * (1.0f / 4096.0f);
  const float var  = r2[0] * (1.0f / 4096.0f) - mean * mean;
  const float rs   = rsqrtf(var + 1e-5f);
#pragma unroll
  for (int j = 0; j < 16; j++) {
    const int idx = tid + j * 256;
    out[(size_t)tok * 4096 + idx] = f2bf((row[idx] - mean) * rs * w[idx] + b[idx]);
  }
}

// ---------------- WMMA GEMM: C[M,N] = A_bf16[M,K] @ W_f32[K,N] + bias ----------------
// Block: 256 thr = 8 waves. Tile: 128(M) x 64(N). Wave w -> rows [w*16, w*16+16), 4 n-tiles.
// EPI 0: f32 store; 1: f32 store + residual; 2: gegelu pair -> bf16 at N/2 columns.
template <int EPI>
__global__ __launch_bounds__(256) void gemm_wmma(const unsigned short* __restrict__ A,
                                                 const float* __restrict__ W,
                                                 const float* __restrict__ bias,
                                                 const float* __restrict__ res,
                                                 void* __restrict__ outp,
                                                 int M, int K, int N) {
  __shared__ unsigned short Bs[64 * 32];   // [n][k], 4 KB
  const int tid = threadIdx.x;
  const int wv = tid >> 5, lane = tid & 31;
  const int bn = blockIdx.x, bm = blockIdx.y;
  const int rowA = bm * 128 + wv * 16 + (lane & 15);
  const unsigned short* arow = A + (size_t)rowA * K;

  v8f acc[4];
#pragma unroll
  for (int i = 0; i < 4; i++)
#pragma unroll
    for (int r = 0; r < 8; r++) acc[i][r] = 0.f;

  const int kl = tid >> 3;            // k row 0..31
  const int n0 = (tid & 7) * 8;       // 8 columns

  for (int kk = 0; kk < K; kk += 32) {
    __syncthreads();
    // stage W tile 32x64 f32 -> bf16 transposed [n][k]
    const float* wp = W + (size_t)(kk + kl) * N + (size_t)bn * 64 + n0;
    const float4 f0 = reinterpret_cast<const float4*>(wp)[0];
    const float4 f1 = reinterpret_cast<const float4*>(wp)[1];
    unsigned short* bsp = Bs + n0 * 32 + kl;
    bsp[0 * 32] = f2bf(f0.x); bsp[1 * 32] = f2bf(f0.y);
    bsp[2 * 32] = f2bf(f0.z); bsp[3 * 32] = f2bf(f0.w);
    bsp[4 * 32] = f2bf(f1.x); bsp[5 * 32] = f2bf(f1.y);
    bsp[6 * 32] = f2bf(f1.z); bsp[7 * 32] = f2bf(f1.w);
    if (kk + 32 < K) __builtin_prefetch(wp + (size_t)32 * N, 0, 0);  // global_prefetch_b8
    __syncthreads();

    const AF a = load_afrag(arow, kk, lane);
#pragma unroll
    for (int nt = 0; nt < 4; nt++) {
      const AF b = load_bfrag(Bs, nt * 16, 32, 0, lane);
      acc[nt] = wmma_bf16(a, b, acc[nt]);
    }
  }

#pragma unroll
  for (int nt = 0; nt < 4; nt++) {
#pragma unroll
    for (int r = 0; r < 8; r++) {
      const int col  = bn * 64 + nt * 16 + (lane & 15);
      const int grow = bm * 128 + wv * 16 + r + ((lane & 16) ? 8 : 0);
      float v = acc[nt][r] + bias[col];
      if (EPI == 0) {
        reinterpret_cast<float*>(outp)[(size_t)grow * N + col] = v;
      } else if (EPI == 1) {
        reinterpret_cast<float*>(outp)[(size_t)grow * N + col] = v + res[(size_t)grow * N + col];
      } else {
        const float partner = __shfl_xor(v, 1);
        if ((lane & 1) == 0) {
          float ag = v, al = partner;
          if (!is_inf(ag)) ag = fminf(ag, 20.0f);
          if (!is_inf(al)) al = fminf(fmaxf(al, -20.0f), 20.0f);
          const float og = ag / (1.0f + __expf(-1.702f * ag));
          reinterpret_cast<unsigned short*>(outp)[(size_t)grow * (N >> 1) + (col >> 1)] =
              f2bf(og * (al + 1.0f));
        }
      }
    }
  }
}

// ---------------- RoPE + head split (f32 qkv -> bf16 q/k/v, head-major) ----------------
__global__ void rope_split(const int* __restrict__ pos, const float* __restrict__ qkv,
                           unsigned short* __restrict__ Qb, unsigned short* __restrict__ Kb,
                           unsigned short* __restrict__ Vb) {
  const int s = blockIdx.x, kv = blockIdx.y, t = threadIdx.x;
  const float* base = qkv + (size_t)s * 6144 + (size_t)kv * 6 * 128;
  const float p = (float)pos[s];
  if (t < 64) {
    const float inv = __expf(-(float)t * (13.815510558f / 64.0f));  // 1e6^(-t/64)
    const float ang = p * inv;
    float sn, c; __sincosf(ang, &sn, &c);
#pragma unroll
    for (int j = 0; j < 5; j++) {
      const float* xp = base + j * 128;
      const float x1 = xp[t], x2 = xp[t + 64];
      const unsigned short lo = f2bf(x1 * c - x2 * sn);
      const unsigned short hi = f2bf(x2 * c + x1 * sn);
      if (j < 4) {
        unsigned short* dq = Qb + ((size_t)(kv * 4 + j) * 2048 + s) * 128;
        dq[t] = lo; dq[t + 64] = hi;
      } else {
        unsigned short* dk = Kb + ((size_t)kv * 2048 + s) * 128;
        dk[t] = lo; dk[t + 64] = hi;
      }
    }
  }
  Vb[((size_t)kv * 2048 + s) * 128 + t] = f2bf(base[5 * 128 + t]);
}

// ---------------- Blocksparse flash attention, full-WMMA ----------------
// Grid (S/128, NH). Block 256 thr = 8 waves; wave w owns query rows [w*16, w*16+16).
__global__ __launch_bounds__(256) void attn_wmma(const unsigned short* __restrict__ Qb,
                                                 const unsigned short* __restrict__ Kb,
                                                 const unsigned short* __restrict__ Vb,
                                                 unsigned short* __restrict__ Ob) {
  __shared__ unsigned short Ks[64 * 128];   // [key][d]      16 KB
  __shared__ unsigned short Vt[128 * 64];   // [d][key]      16 KB
  __shared__ unsigned short Ps[128 * 64];   // [q][key] bf16 16 KB
  const int qt = blockIdx.x, h = blockIdx.y;
  const int tid = threadIdx.x, wv = tid >> 5, lane = tid & 31;
  const int kvh = h >> 2;                   // NQ = 4
  const int rowq = qt * 128 + wv * 16 + (lane & 15);
  const unsigned short* qrow = Qb + ((size_t)h * 2048 + rowq) * 128;

  v8f o[8];
#pragma unroll
  for (int i = 0; i < 8; i++)
#pragma unroll
    for (int r = 0; r < 8; r++) o[i][r] = 0.f;
  float mi[8], li[8];
#pragma unroll
  for (int r = 0; r < 8; r++) { mi[r] = -1e30f; li[r] = 0.f; }

  const int qb1 = qt * 2 + 1;
  const float scale = 0.08838834764831845f;  // 1/sqrt(128)

  for (int kb = 0; kb <= qb1; ++kb) {
    const bool vert = ((kb + 1) & 7) == 0;                 // (kb+1) % VERT == 0
    if (!(((qb1 - kb) < 16) || vert)) continue;            // block-level skip
    __syncthreads();
    { // stage K [key][d] and V transposed [d][key]
      const int r = tid >> 2, c0 = (tid & 3) * 32;
      const uint4* ks = reinterpret_cast<const uint4*>(
          Kb + ((size_t)kvh * 2048 + kb * 64 + r) * 128 + c0);
      uint4* kd = reinterpret_cast<uint4*>(Ks + r * 128 + c0);
#pragma unroll
      for (int i = 0; i < 4; i++) kd[i] = ks[i];
      const uint4* vs = reinterpret_cast<const uint4*>(
          Vb + ((size_t)kvh * 2048 + kb * 64 + r) * 128 + c0);
#pragma unroll
      for (int i = 0; i < 4; i++) {
        union { uint4 q; unsigned short e[8]; } u; u.q = vs[i];
#pragma unroll
        for (int j = 0; j < 8; j++) Vt[(c0 + i * 8 + j) * 64 + r] = u.e[j];
      }
    }
    __syncthreads();

    // S = Q @ K^T  (d loop: 128 = 4 x 32)
    v8f sacc[4];
#pragma unroll
    for (int i = 0; i < 4; i++)
#pragma unroll
      for (int r = 0; r < 8; r++) sacc[i][r] = 0.f;
#pragma unroll
    for (int d4 = 0; d4 < 4; d4++) {
      const AF a = load_afrag(qrow, d4 * 32, lane);
#pragma unroll
      for (int nt = 0; nt < 4; nt++) {
        const AF b = load_bfrag(Ks, nt * 16, 128, d4 * 32, lane);
        sacc[nt] = wmma_bf16(a, b, sacc[nt]);
      }
    }

    // scale + element mask (causal + local/vertical per query block)
#pragma unroll
    for (int nt = 0; nt < 4; nt++)
#pragma unroll
      for (int r = 0; r < 8; r++) {
        const int gq = qt * 128 + wv * 16 + r + ((lane & 16) ? 8 : 0);
        const int gk = kb * 64 + nt * 16 + (lane & 15);
        const float s = sacc[nt][r] * scale;
        const bool ok = (gk <= gq) && ((((gq >> 6) - kb) < 16) || vert);
        sacc[nt][r] = ok ? s : -1e30f;
      }

    // online softmax (rows live in 16-lane halves; xor-shuffles stay in-half)
    float corr[8];
#pragma unroll
    for (int r = 0; r < 8; r++) {
      float rm = fmaxf(fmaxf(sacc[0][r], sacc[1][r]), fmaxf(sacc[2][r], sacc[3][r]));
      rm = fmaxf(rm, __shfl_xor(rm, 1));
      rm = fmaxf(rm, __shfl_xor(rm, 2));
      rm = fmaxf(rm, __shfl_xor(rm, 4));
      rm = fmaxf(rm, __shfl_xor(rm, 8));
      const float mn = fmaxf(mi[r], rm);
      corr[r] = __expf(mi[r] - mn);
      mi[r] = mn;
    }
#pragma unroll
    for (int nt = 0; nt < 4; nt++)
#pragma unroll
      for (int r = 0; r < 8; r++) sacc[nt][r] = __expf(sacc[nt][r] - mi[r]);
#pragma unroll
    for (int r = 0; r < 8; r++) {
      float rs = sacc[0][r] + sacc[1][r] + sacc[2][r] + sacc[3][r];
      rs += __shfl_xor(rs, 1);
      rs += __shfl_xor(rs, 2);
      rs += __shfl_xor(rs, 4);
      rs += __shfl_xor(rs, 8);
      li[r] = li[r] * corr[r] + rs;
#pragma unroll
      for (int dt = 0; dt < 8; dt++) o[dt][r] *= corr[r];
    }

    // P -> LDS (wave-private strip; same-wave DS ops are in-order)
#pragma unroll
    for (int nt = 0; nt < 4; nt++)
#pragma unroll
      for (int r = 0; r < 8; r++) {
        const int prow = wv * 16 + r + ((lane & 16) ? 8 : 0);
        Ps[prow * 64 + nt * 16 + (lane & 15)] = f2bf(sacc[nt][r]);
      }

    // O += P @ V  (key loop: 64 = 2 x 32)
    const unsigned short* prow_ptr = Ps + (size_t)(wv * 16 + (lane & 15)) * 64;
#pragma unroll
    for (int kc = 0; kc < 2; kc++) {
      const AF pa = load_afrag(prow_ptr, kc * 32, lane);
#pragma unroll
      for (int dt = 0; dt < 8; dt++) {
        const AF vb = load_bfrag(Vt, dt * 16, 64, kc * 32, lane);
        o[dt] = wmma_bf16(pa, vb, o[dt]);
      }
    }
  }

  // normalize + store token-major bf16 [S][NH*HD]
#pragma unroll
  for (int r = 0; r < 8; r++) {
    const float inv = 1.0f / li[r];
    const int gq = qt * 128 + wv * 16 + r + ((lane & 16) ? 8 : 0);
#pragma unroll
    for (int dt = 0; dt < 8; dt++)
      Ob[(size_t)gq * 4096 + h * 128 + dt * 16 + (lane & 15)] = f2bf(o[dt][r] * inv);
  }
}

// ---------------- Orchestration ----------------
extern "C" void kernel_launch(void* const* d_in, const int* in_sizes, int n_in,
                              void* d_out, int out_size, void* d_ws, size_t ws_size,
                              hipStream_t stream) {
  const int*   positions = (const int*)d_in[0];
  const float* hs    = (const float*)d_in[1];
  const float* ln1w  = (const float*)d_in[2];
  const float* ln1b  = (const float*)d_in[3];
  const float* Wqkv  = (const float*)d_in[4];
  const float* bqkv  = (const float*)d_in[5];
  const float* Wo    = (const float*)d_in[6];
  const float* bo    = (const float*)d_in[7];
  const float* ln2w  = (const float*)d_in[8];
  const float* ln2b  = (const float*)d_in[9];
  const float* Wup   = (const float*)d_in[10];
  const float* bup   = (const float*)d_in[11];
  const float* Wdn   = (const float*)d_in[12];
  const float* bdn   = (const float*)d_in[13];
  float* out = (float*)d_out;

  char* ws = (char*)d_ws;
  size_t off = 0;
  unsigned short* xnorm = (unsigned short*)(ws + off); off += (size_t)2048 * 4096 * 2;
  float*          qkv   = (float*)(ws + off);          off += (size_t)2048 * 6144 * 4;
  unsigned short* qbf   = (unsigned short*)(ws + off); off += (size_t)32 * 2048 * 128 * 2;
  unsigned short* kbf   = (unsigned short*)(ws + off); off += (size_t)8 * 2048 * 128 * 2;
  unsigned short* vbf   = (unsigned short*)(ws + off); off += (size_t)8 * 2048 * 128 * 2;
  unsigned short* attnb = (unsigned short*)(ws + off); off += (size_t)2048 * 4096 * 2;
  float*          hid   = (float*)(ws + off);          off += (size_t)2048 * 4096 * 4;
  unsigned short* gbuf  = (unsigned short*)(ws + off); off += (size_t)2048 * 14336 * 2;

  // LN1 -> bf16
  ln_bf16<<<2048, 256, 0, stream>>>(hs, ln1w, ln1b, xnorm);
  // QKV: [2048,4096] @ [4096,6144]
  gemm_wmma<0><<<dim3(96, 16), 256, 0, stream>>>(xnorm, Wqkv, bqkv, nullptr, qkv,
                                                 2048, 4096, 6144);
  // RoPE + split to head-major bf16
  rope_split<<<dim3(2048, 8), 128, 0, stream>>>(positions, qkv, qbf, kbf, vbf);
  // Blocksparse flash attention
  attn_wmma<<<dim3(16, 32), 256, 0, stream>>>(qbf, kbf, vbf, attnb);
  // Wo + residual -> hidden f32
  gemm_wmma<1><<<dim3(64, 16), 256, 0, stream>>>(attnb, Wo, bo, hs, hid,
                                                 2048, 4096, 4096);
  // LN2 -> bf16
  ln_bf16<<<2048, 256, 0, stream>>>(hid, ln2w, ln2b, xnorm);
  // Up + fused gegelu -> bf16 [2048,14336]
  gemm_wmma<2><<<dim3(448, 16), 256, 0, stream>>>(xnorm, Wup, bup, nullptr, gbuf,
                                                  2048, 4096, 28672);
  // Down + residual -> final f32 output
  gemm_wmma<1><<<dim3(64, 16), 256, 0, stream>>>(gbuf, Wdn, bdn, hid, out,
                                                 2048, 14336, 4096);
}
